// NeptuneTransformerEncoderLayer_30210799960116
// MI455X (gfx1250) — matrix operand order repaired and checked by reading, not compiled, not run on hardware
//
#include <hip/hip_runtime.h>
#include <hip/hip_bf16.h>

typedef __attribute__((ext_vector_type(16))) _Float16 v16h;
typedef __attribute__((ext_vector_type(8)))  _Float16 v8h;
typedef __attribute__((ext_vector_type(8)))  float    v8f;
typedef __attribute__((ext_vector_type(4)))  int      v4i;

#define D_MODEL  1024
#define NHEAD    16
#define HEAD_DIM 64
#define DFF      4096
#define BATCH    4
#define SEQ      2048
#define MTOT     (BATCH * SEQ)   // 8192
#define EPS_RMS  1e-5f

// ---------------------------------------------------------------------------
// CDNA5 async global->LDS path (ASYNCcnt-tracked), guarded so the file still
// compiles on toolchains without the builtin.  Probe round 2 showed the
// builtin takes generic `int4*` data pointers.
// ---------------------------------------------------------------------------
#if defined(__has_builtin)
#if __has_builtin(__builtin_amdgcn_global_load_async_to_lds_b128) && \
    __has_builtin(__builtin_amdgcn_s_wait_asynccnt)
#define USE_ASYNC_LDS 1
#endif
#endif

#ifdef USE_ASYNC_LDS
__device__ __forceinline__ void async_copy_b128(const _Float16* g, _Float16* l) {
  __builtin_amdgcn_global_load_async_to_lds_b128((v4i*)g, (v4i*)l, 0, 0);
}
#endif

// ---------------------------------------------------------------------------
// WMMA helpers (gfx1250, wave32)
// ---------------------------------------------------------------------------
__device__ __forceinline__ v8f wmma_f16(v16h a, v16h b, v8f c) {
  return __builtin_amdgcn_wmma_f32_16x16x32_f16(
      /*neg_a=*/false, a, /*neg_b=*/false, b,
      /*c_mod=*/(short)0, c, /*reuse_a=*/false, /*reuse_b=*/false);
}

// A-matrix 16x32 f16 fragment from row-major LDS tile (lda in halves).
// Lane L holds row M=L%16; half-wave hi=L/16 holds K = hi*16 .. hi*16+15
// (same contiguous K order as load_b_frag).
__device__ __forceinline__ v16h load_a_frag(const _Float16* a, int lda) {
  const int lane = threadIdx.x & 31;
  const int m  = lane & 15;
  const int hi = lane >> 4;
  const _Float16* p = a + m * lda + hi * 16;
  v8h lo = *(const v8h*)(p);
  v8h h2 = *(const v8h*)(p + 8);
  v16h r;
#pragma unroll
  for (int j = 0; j < 8; ++j) { r[j] = lo[j]; r[8 + j] = h2[j]; }
  return r;
}

// B-matrix 32x16 f16 fragment from N-major (Bt) LDS tile: bt[n][k], ldb halves.
// Lane L holds column N=L%16; half hi=L/16 holds K = hi*16 .. hi*16+15.
__device__ __forceinline__ v16h load_b_frag(const _Float16* bt, int ldb) {
  const int lane = threadIdx.x & 31;
  const int n  = lane & 15;
  const int hi = lane >> 4;
  return *(const v16h*)(bt + n * ldb + hi * 16);
}

// ---------------------------------------------------------------------------
// Elementwise conversion: fp32 -> fp16
// ---------------------------------------------------------------------------
__global__ void f32_to_f16_kernel(const float* __restrict__ src,
                                  _Float16* __restrict__ dst, int n) {
  int i = blockIdx.x * 256 + threadIdx.x;
  if (i < n) dst[i] = (_Float16)src[i];
}

// ---------------------------------------------------------------------------
// RMSNorm: one block per row of 1024, output f16
// ---------------------------------------------------------------------------
__global__ __launch_bounds__(256)
void rmsnorm_kernel(const float* __restrict__ x, const float* __restrict__ w,
                    _Float16* __restrict__ out) {
  __shared__ float red[8];
  const int row = blockIdx.x;
  const float* xr = x + (size_t)row * D_MODEL;
  float ss = 0.f;
#pragma unroll
  for (int j = threadIdx.x; j < D_MODEL; j += 256) {
    float v = xr[j];
    ss += v * v;
  }
#pragma unroll
  for (int off = 16; off; off >>= 1) ss += __shfl_xor(ss, off, 32);
  if ((threadIdx.x & 31) == 0) red[threadIdx.x >> 5] = ss;
  __syncthreads();
  if (threadIdx.x == 0) {
    float t = 0.f;
#pragma unroll
    for (int i = 0; i < 8; ++i) t += red[i];
    red[0] = rsqrtf(t * (1.0f / (float)D_MODEL) + EPS_RMS);
  }
  __syncthreads();
  const float rn = red[0];
#pragma unroll
  for (int j = threadIdx.x; j < D_MODEL; j += 256)
    out[(size_t)row * D_MODEL + j] = (_Float16)(xr[j] * rn * w[j]);
}

// ---------------------------------------------------------------------------
// Generic GEMM: D[M,N] = A[M,K] (f16, row-major) x Bt[N,K] (f16, row-major)^T
// Block tile 128x128, 8 waves (4x2), wave tile 32x64, K-step 32.
// Double-buffered LDS; next tile staged (async-to-LDS when available) while
// the current tile runs through WMMA.  One barrier per K-step.
// EPI==0: store f16.  EPI==1: store f32 with residual add.
// ---------------------------------------------------------------------------
template <int EPI>
__global__ __launch_bounds__(256)
void gemm_kernel(const _Float16* __restrict__ A, const _Float16* __restrict__ Bt,
                 const float* __restrict__ resid, _Float16* __restrict__ Dh,
                 float* __restrict__ Df, int M, int N, int K) {
  __shared__ __align__(32) _Float16 As[2][128 * 48];
  __shared__ __align__(32) _Float16 Bs[2][128 * 48];
  const int tid  = threadIdx.x;
  const int lane = tid & 31, wid = tid >> 5;
  const int wm = wid >> 1, wn = wid & 1;
  const int bm = blockIdx.y * 128, bn = blockIdx.x * 128;

  const int row = tid >> 1, seg = tid & 1;
  const _Float16* gA = &A[(size_t)(bm + row) * K + seg * 16];
  const _Float16* gB = &Bt[(size_t)(bn + row) * K + seg * 16];
  const int ldsOff = row * 48 + seg * 16;

  v8f acc[2][4];
#pragma unroll
  for (int i = 0; i < 2; ++i)
#pragma unroll
    for (int j = 0; j < 4; ++j) acc[i][j] = (v8f){0, 0, 0, 0, 0, 0, 0, 0};

  const int nk = K >> 5;

  // Prologue: stage tile 0 into buffer 0.
#ifdef USE_ASYNC_LDS
  async_copy_b128(gA, &As[0][ldsOff]);
  async_copy_b128(gB, &Bs[0][ldsOff]);
  __builtin_amdgcn_s_wait_asynccnt(0);
#else
  *(v16h*)&As[0][ldsOff] = *(const v16h*)gA;
  *(v16h*)&Bs[0][ldsOff] = *(const v16h*)gB;
#endif
  __syncthreads();

  for (int t = 0; t < nk; ++t) {
    const int cur = t & 1, nxt = cur ^ 1;

#ifdef USE_ASYNC_LDS
    if (t + 1 < nk) {
      async_copy_b128(gA + (size_t)(t + 1) * 32, &As[nxt][ldsOff]);
      async_copy_b128(gB + (size_t)(t + 1) * 32, &Bs[nxt][ldsOff]);
    }
#else
    v16h ra, rb;
    if (t + 1 < nk) {
      ra = *(const v16h*)(gA + (size_t)(t + 1) * 32);
      rb = *(const v16h*)(gB + (size_t)(t + 1) * 32);
    }
#endif
    if (t + 2 < nk) {
      __builtin_prefetch(gA + (size_t)(t + 2) * 32, 0, 1);
      __builtin_prefetch(gB + (size_t)(t + 2) * 32, 0, 1);
    }

    // Compute on the current tile (overlaps in-flight staging loads).
    v16h af[2], bf[4];
#pragma unroll
    for (int im = 0; im < 2; ++im)
      af[im] = load_a_frag(&As[cur][(wm * 32 + im * 16) * 48], 48);
#pragma unroll
    for (int jn = 0; jn < 4; ++jn)
      bf[jn] = load_b_frag(&Bs[cur][(wn * 64 + jn * 16) * 48], 48);
#pragma unroll
    for (int im = 0; im < 2; ++im)
#pragma unroll
      for (int jn = 0; jn < 4; ++jn)
        acc[im][jn] = wmma_f16(af[im], bf[jn], acc[im][jn]);

#ifdef USE_ASYNC_LDS
    if (t + 1 < nk) __builtin_amdgcn_s_wait_asynccnt(0);
#else
    if (t + 1 < nk) {
      *(v16h*)&As[nxt][ldsOff] = ra;
      *(v16h*)&Bs[nxt][ldsOff] = rb;
    }
#endif
    __syncthreads();
  }

  // Epilogue. C layout: VGPR i -> row = i + (lane>=16 ? 8 : 0), col = lane%16.
  const int hi = lane >> 4, col = lane & 15;
#pragma unroll
  for (int im = 0; im < 2; ++im)
#pragma unroll
    for (int jn = 0; jn < 4; ++jn)
#pragma unroll
      for (int i = 0; i < 8; ++i) {
        const int gm = bm + wm * 32 + im * 16 + i + hi * 8;
        const int gn = bn + wn * 64 + jn * 16 + col;
        const size_t idx = (size_t)gm * N + gn;
        if (EPI == 0) {
          Dh[idx] = (_Float16)acc[im][jn][i];
        } else {
          Df[idx] = resid[idx] + acc[im][jn][i];
        }
      }
}

// ---------------------------------------------------------------------------
// Fused dual GEMM with SiLU-gate epilogue:
//   D = silu(A x B1t^T) * (A x B3t^T), f16 out.
// Block tile 128x64, 8 waves (4x2), wave tile 32x32 (x2 accumulator sets).
// Same double-buffered pipeline as gemm_kernel.
// ---------------------------------------------------------------------------
__global__ __launch_bounds__(256)
void gemm_dual_silu_kernel(const _Float16* __restrict__ A,
                           const _Float16* __restrict__ B1t,
                           const _Float16* __restrict__ B3t,
                           _Float16* __restrict__ Dh, int M, int N, int K) {
  __shared__ __align__(32) _Float16 As[2][128 * 48];
  __shared__ __align__(32) _Float16 B1s[2][64 * 48];
  __shared__ __align__(32) _Float16 B3s[2][64 * 48];
  const int tid  = threadIdx.x;
  const int lane = tid & 31, wid = tid >> 5;
  const int wm = wid >> 1, wn = wid & 1;
  const int bm = blockIdx.y * 128, bn = blockIdx.x * 64;

  const int rowA = tid >> 1, segA = tid & 1;
  const _Float16* gA = &A[(size_t)(bm + rowA) * K + segA * 16];
  const int ldsOffA = rowA * 48 + segA * 16;

  const int t2 = tid & 127;
  const int rowB = t2 >> 1, segB = t2 & 1;
  const _Float16* gB = ((tid < 128) ? B1t : B3t) + (size_t)(bn + rowB) * K + segB * 16;
  const int ldsOffB = rowB * 48 + segB * 16;

  v8f a1[2][2], a3[2][2];
#pragma unroll
  for (int i = 0; i < 2; ++i)
#pragma unroll
    for (int j = 0; j < 2; ++j) {
      a1[i][j] = (v8f){0, 0, 0, 0, 0, 0, 0, 0};
      a3[i][j] = (v8f){0, 0, 0, 0, 0, 0, 0, 0};
    }

  const int nk = K >> 5;

#ifdef USE_ASYNC_LDS
  async_copy_b128(gA, &As[0][ldsOffA]);
  async_copy_b128(gB, (tid < 128) ? &B1s[0][ldsOffB] : &B3s[0][ldsOffB]);
  __builtin_amdgcn_s_wait_asynccnt(0);
#else
  *(v16h*)&As[0][ldsOffA] = *(const v16h*)gA;
  *(v16h*)((tid < 128) ? &B1s[0][ldsOffB] : &B3s[0][ldsOffB]) = *(const v16h*)gB;
#endif
  __syncthreads();

  for (int t = 0; t < nk; ++t) {
    const int cur = t & 1, nxt = cur ^ 1;

#ifdef USE_ASYNC_LDS
    if (t + 1 < nk) {
      async_copy_b128(gA + (size_t)(t + 1) * 32, &As[nxt][ldsOffA]);
      async_copy_b128(gB + (size_t)(t + 1) * 32,
                      (tid < 128) ? &B1s[nxt][ldsOffB] : &B3s[nxt][ldsOffB]);
    }
#else
    v16h ra, rb;
    if (t + 1 < nk) {
      ra = *(const v16h*)(gA + (size_t)(t + 1) * 32);
      rb = *(const v16h*)(gB + (size_t)(t + 1) * 32);
    }
#endif
    if (t + 2 < nk) {
      __builtin_prefetch(gA + (size_t)(t + 2) * 32, 0, 1);
      __builtin_prefetch(gB + (size_t)(t + 2) * 32, 0, 1);
    }

    v16h af[2], b1f[2], b3f[2];
#pragma unroll
    for (int im = 0; im < 2; ++im)
      af[im] = load_a_frag(&As[cur][(wm * 32 + im * 16) * 48], 48);
#pragma unroll
    for (int jn = 0; jn < 2; ++jn) {
      b1f[jn] = load_b_frag(&B1s[cur][(wn * 32 + jn * 16) * 48], 48);
      b3f[jn] = load_b_frag(&B3s[cur][(wn * 32 + jn * 16) * 48], 48);
    }
#pragma unroll
    for (int im = 0; im < 2; ++im)
#pragma unroll
      for (int jn = 0; jn < 2; ++jn) {
        a1[im][jn] = wmma_f16(af[im], b1f[jn], a1[im][jn]);
        a3[im][jn] = wmma_f16(af[im], b3f[jn], a3[im][jn]);
      }

#ifdef USE_ASYNC_LDS
    if (t + 1 < nk) __builtin_amdgcn_s_wait_asynccnt(0);
#else
    if (t + 1 < nk) {
      *(v16h*)&As[nxt][ldsOffA] = ra;
      *(v16h*)((tid < 128) ? &B1s[nxt][ldsOffB] : &B3s[nxt][ldsOffB]) = rb;
    }
#endif
    __syncthreads();
  }

  const int hi = lane >> 4, col = lane & 15;
#pragma unroll
  for (int im = 0; im < 2; ++im)
#pragma unroll
    for (int jn = 0; jn < 2; ++jn)
#pragma unroll
      for (int i = 0; i < 8; ++i) {
        const int gm = bm + wm * 32 + im * 16 + i + hi * 8;
        const int gn = bn + wn * 32 + jn * 16 + col;
        const float g = a1[im][jn][i];
        const float s = g * __builtin_amdgcn_rcpf(1.f + __expf(-g));
        Dh[(size_t)gm * N + gn] = (_Float16)(s * a3[im][jn][i]);
      }
}

// ---------------------------------------------------------------------------
// RoPE (4 axes x 8 interleaved pairs) in place on f16 q and k inside qkv
// ---------------------------------------------------------------------------
__global__ __launch_bounds__(256)
void rope_kernel(_Float16* __restrict__ qkv, const float* __restrict__ positions,
                 const float* __restrict__ axis_scale) {
  const int idx = blockIdx.x * 256 + threadIdx.x;  // (bs, h, axis)
  const int a  = idx & 3;
  const int h  = (idx >> 2) & 15;
  const int bs = idx >> 6;
  if (bs >= MTOT) return;
  const float coord = positions[bs * 4 + a] * axis_scale[a];
  _Float16* qp = qkv + (size_t)bs * (3 * D_MODEL) + h * HEAD_DIM + a * 16;
  _Float16* kp = qp + D_MODEL;
  const float NEG_LN_BASE_OVER_8 = -9.210340371976184f / 8.0f;  // -ln(10000)/8
#pragma unroll
  for (int j = 0; j < 8; ++j) {
    const float inv = __expf((float)j * NEG_LN_BASE_OVER_8);
    const float ang = coord * inv;
    const float c = __cosf(ang), s = __sinf(ang);
    float e = (float)qp[2 * j], o = (float)qp[2 * j + 1];
    qp[2 * j]     = (_Float16)(e * c - o * s);
    qp[2 * j + 1] = (_Float16)(o * c + e * s);
    e = (float)kp[2 * j];
    o = (float)kp[2 * j + 1];
    kp[2 * j]     = (_Float16)(e * c - o * s);
    kp[2 * j + 1] = (_Float16)(o * c + e * s);
  }
}

// ---------------------------------------------------------------------------
// Flash attention: grid (S/64, B*H), 4 waves/block, each wave owns 16 query
// rows.  QK^T and P*V via WMMA, online softmax through LDS.
// ---------------------------------------------------------------------------
__global__ __launch_bounds__(128)
void attn_kernel(const _Float16* __restrict__ qkv,
                 _Float16* __restrict__ attn_out) {
  const int bh = blockIdx.y;
  const int bb = bh >> 4, h = bh & 15;
  const int q0 = blockIdx.x * 64;
  const int tid = threadIdx.x, lane = tid & 31, wid = tid >> 5;

  __shared__ __align__(32) _Float16 Qs[64 * 80];
  __shared__ __align__(32) _Float16 Ks[64 * 80];
  __shared__ __align__(32) _Float16 Vts[64 * 80];   // Vt[d][k]
  __shared__ float   Ss[4][16 * 64];
  __shared__ __align__(32) _Float16 Ps[4][16 * 80];
  __shared__ float m_s[64], l_s[64], sc_s[64];

  const size_t rs = 3 * D_MODEL;  // qkv row stride in halves
  const _Float16* qbase = qkv + (size_t)bb * SEQ * rs + h * HEAD_DIM;

  // Stage Q (pre-scaled by 1/sqrt(64) = 0.125)
  {
    const int r = tid >> 1, seg = tid & 1;
    const _Float16* src = qbase + (size_t)(q0 + r) * rs + seg * 32;
#pragma unroll
    for (int j = 0; j < 32; ++j)
      Qs[r * 80 + seg * 32 + j] = (_Float16)(0.125f * (float)src[j]);
  }
  if (tid < 64) { m_s[tid] = -1e30f; l_s[tid] = 0.f; }
  __syncthreads();

  v8f o[4];
#pragma unroll
  for (int j = 0; j < 4; ++j) o[j] = (v8f){0, 0, 0, 0, 0, 0, 0, 0};

  for (int kb = 0; kb < SEQ; kb += 64) {
    // Stage K (row-major) and V (transposed)
    {
      const int r = tid >> 1, seg = tid & 1;
      const _Float16* ksrc = qbase + D_MODEL + (size_t)(kb + r) * rs + seg * 32;
      *(v16h*)&Ks[r * 80 + seg * 32]      = *(const v16h*)(ksrc);
      *(v16h*)&Ks[r * 80 + seg * 32 + 16] = *(const v16h*)(ksrc + 16);
      const _Float16* vsrc = qbase + 2 * D_MODEL + (size_t)(kb + r) * rs + seg * 32;
#pragma unroll
      for (int j = 0; j < 32; ++j) Vts[(seg * 32 + j) * 80 + r] = vsrc[j];
    }
    __syncthreads();

    // Scores: S = Q K^T  (wave rows [wid*16, wid*16+16), 64 key cols)
    v8f sfr[4];
#pragma unroll
    for (int j = 0; j < 4; ++j) sfr[j] = (v8f){0, 0, 0, 0, 0, 0, 0, 0};
#pragma unroll
    for (int jn = 0; jn < 4; ++jn)
#pragma unroll
      for (int kk = 0; kk < 2; ++kk) {
        v16h af  = load_a_frag(&Qs[(wid * 16) * 80 + kk * 32], 80);
        v16h bfr = load_b_frag(&Ks[(jn * 16) * 80 + kk * 32], 80);
        sfr[jn] = wmma_f16(af, bfr, sfr[jn]);
      }
    {
      const int hi = lane >> 4, col = lane & 15;
#pragma unroll
      for (int jn = 0; jn < 4; ++jn)
#pragma unroll
        for (int i = 0; i < 8; ++i)
          Ss[wid][(i + hi * 8) * 64 + jn * 16 + col] = sfr[jn][i];
    }
    __syncthreads();

    // Online softmax: lanes 0..15 of each wave own one row each
    if (lane < 16) {
      const int r = wid * 16 + lane;
      const float* srow = &Ss[wid][lane * 64];
      float mloc = -1e30f;
#pragma unroll
      for (int j = 0; j < 64; ++j) mloc = fmaxf(mloc, srow[j]);
      const float mold = m_s[r];
      const float mnew = fmaxf(mold, mloc);
      const float scale = __expf(mold - mnew);
      float lsum = 0.f;
      _Float16* prow = &Ps[wid][lane * 80];
#pragma unroll
      for (int j = 0; j < 64; ++j) {
        const float p = __expf(srow[j] - mnew);
        lsum += p;
        prow[j] = (_Float16)p;
      }
      m_s[r] = mnew;
      l_s[r] = l_s[r] * scale + lsum;
      sc_s[r] = scale;
    }
    __syncthreads();

    // Rescale running output, then O += P V
    {
      const int hi = lane >> 4;
#pragma unroll
      for (int jn = 0; jn < 4; ++jn)
#pragma unroll
        for (int i = 0; i < 8; ++i)
          o[jn][i] *= sc_s[wid * 16 + i + hi * 8];
    }
#pragma unroll
    for (int jn = 0; jn < 4; ++jn)
#pragma unroll
      for (int kk = 0; kk < 2; ++kk) {
        v16h af  = load_a_frag(&Ps[wid][kk * 32], 80);
        v16h bfr = load_b_frag(&Vts[(jn * 16) * 80 + kk * 32], 80);
        o[jn] = wmma_f16(af, bfr, o[jn]);
      }
    __syncthreads();
  }

  // Normalize and store to (b, s, h*64+d) layout for the out-projection GEMM
  {
    const int hi = lane >> 4, col = lane & 15;
#pragma unroll
    for (int jn = 0; jn < 4; ++jn)
#pragma unroll
      for (int i = 0; i < 8; ++i) {
        const int r = wid * 16 + i + hi * 8;
        const int s = q0 + r;
        const float inv = __builtin_amdgcn_rcpf(l_s[r]);
        attn_out[((size_t)(bb * SEQ + s)) * D_MODEL + h * HEAD_DIM + jn * 16 + col] =
            (_Float16)(o[jn][i] * inv);
      }
  }
}

// ---------------------------------------------------------------------------
// Host-side orchestration
// ---------------------------------------------------------------------------
extern "C" void kernel_launch(void* const* d_in, const int* in_sizes, int n_in,
                              void* d_out, int out_size, void* d_ws, size_t ws_size,
                              hipStream_t stream) {
  const float* src        = (const float*)d_in[0];
  const float* positions  = (const float*)d_in[1];
  const float* w_qkv      = (const float*)d_in[2];
  const float* w_out      = (const float*)d_in[3];
  const float* norm1_w    = (const float*)d_in[4];
  const float* norm2_w    = (const float*)d_in[5];
  const float* w1         = (const float*)d_in[6];
  const float* w2         = (const float*)d_in[7];
  const float* w3         = (const float*)d_in[8];
  const float* axis_scale = (const float*)d_in[9];
  float* out = (float*)d_out;

  // Workspace layout (bytes), ffn hidden aliases dead qkv+attn region.
  char* ws = (char*)d_ws;
  _Float16* wqkv_h = (_Float16*)ws;  ws += (size_t)3 * D_MODEL * D_MODEL * 2;
  _Float16* wout_h = (_Float16*)ws;  ws += (size_t)D_MODEL * D_MODEL * 2;
  _Float16* w1_h   = (_Float16*)ws;  ws += (size_t)DFF * D_MODEL * 2;
  _Float16* w3_h   = (_Float16*)ws;  ws += (size_t)DFF * D_MODEL * 2;
  _Float16* w2_h   = (_Float16*)ws;  ws += (size_t)D_MODEL * DFF * 2;
  _Float16* xnorm_h = (_Float16*)ws; ws += (size_t)MTOT * D_MODEL * 2;
  _Float16* qkv_h   = (_Float16*)ws; ws += (size_t)MTOT * 3 * D_MODEL * 2;
  _Float16* attn_h  = (_Float16*)ws; ws += (size_t)MTOT * D_MODEL * 2;
  float*    x1      = (float*)ws;    ws += (size_t)MTOT * D_MODEL * 4;
  _Float16* ffh_h   = qkv_h;  // MTOT x 4096 halves == qkv(3072) + attn(1024) region

  // 1) Convert weights to f16
  {
    const int n1 = 3 * D_MODEL * D_MODEL;
    f32_to_f16_kernel<<<(n1 + 255) / 256, 256, 0, stream>>>(w_qkv, wqkv_h, n1);
    const int n2 = D_MODEL * D_MODEL;
    f32_to_f16_kernel<<<(n2 + 255) / 256, 256, 0, stream>>>(w_out, wout_h, n2);
    const int n3 = DFF * D_MODEL;
    f32_to_f16_kernel<<<(n3 + 255) / 256, 256, 0, stream>>>(w1, w1_h, n3);
    f32_to_f16_kernel<<<(n3 + 255) / 256, 256, 0, stream>>>(w3, w3_h, n3);
    f32_to_f16_kernel<<<(n3 + 255) / 256, 256, 0, stream>>>(w2, w2_h, n3);
  }

  // 2) RMSNorm 1
  rmsnorm_kernel<<<MTOT, 256, 0, stream>>>(src, norm1_w, xnorm_h);

  // 3) QKV projection: (8192 x 1024) x (3072 x 1024)^T -> f16
  gemm_kernel<0><<<dim3(3 * D_MODEL / 128, MTOT / 128), 256, 0, stream>>>(
      xnorm_h, wqkv_h, nullptr, qkv_h, nullptr, MTOT, 3 * D_MODEL, D_MODEL);

  // 4) RoPE in place on q,k
  rope_kernel<<<(MTOT * NHEAD * 4) / 256, 256, 0, stream>>>(qkv_h, positions,
                                                            axis_scale);

  // 5) Attention
  attn_kernel<<<dim3(SEQ / 64, BATCH * NHEAD), 128, 0, stream>>>(qkv_h, attn_h);

  // 6) Output projection + residual -> x1 (f32)
  gemm_kernel<1><<<dim3(D_MODEL / 128, MTOT / 128), 256, 0, stream>>>(
      attn_h, wout_h, src, nullptr, x1, MTOT, D_MODEL, D_MODEL);

  // 7) RMSNorm 2 (reuse xnorm_h)
  rmsnorm_kernel<<<MTOT, 256, 0, stream>>>(x1, norm2_w, xnorm_h);

  // 8) Fused gate GEMM: silu(x w1^T) * (x w3^T) -> f16
  gemm_dual_silu_kernel<<<dim3(DFF / 64, MTOT / 128), 256, 0, stream>>>(
      xnorm_h, w1_h, w3_h, ffh_h, MTOT, DFF, D_MODEL);

  // 9) FFN down-proj + residual -> d_out (f32)
  gemm_kernel<1><<<dim3(D_MODEL / 128, MTOT / 128), 256, 0, stream>>>(
      ffh_h, w2_h, x1, nullptr, out, MTOT, D_MODEL, DFF);
}
